// Model_38577396252772
// MI455X (gfx1250) — compile-verified, matrix operand order checked by reading, not run on hardware
//
#include <hip/hip_runtime.h>
#include <hip/hip_bf16.h>
#include <math.h>

typedef _Float16 f16;
typedef _Float16 v16h __attribute__((ext_vector_type(16)));
typedef _Float16 v8h  __attribute__((ext_vector_type(8)));
typedef float    v8f  __attribute__((ext_vector_type(8)));

#define DEV static __device__ __forceinline__

#ifndef __has_builtin
#define __has_builtin(x) 0
#endif
#if __has_builtin(__builtin_amdgcn_global_load_async_to_lds_b128) && \
    __has_builtin(__builtin_amdgcn_s_wait_asynccnt)
#define USE_ASYNC_LDS 1
#else
#define USE_ASYNC_LDS 0
#endif

#if USE_ASYNC_LDS
// Builtin signature (from hipcc diagnostic): first param is
// 'int __vector(4) __device__ *' i.e. GCC vector_size(16) int in AS1 (global);
// second is the LDS destination in AS3; then i32 offset, i32 cpol.
typedef int i4v __attribute__((vector_size(16)));
typedef __attribute__((address_space(1))) i4v g_i4v;
typedef __attribute__((address_space(3))) i4v l_i4v;
DEV void async_cp16(const f16* g, f16* l) {
  // one b128 per lane: LDS[l] = MEM[g]; tracked with ASYNCcnt
  __builtin_amdgcn_global_load_async_to_lds_b128((g_i4v*)g, (l_i4v*)l, 0, 0);
}
DEV void async_wait0() { __builtin_amdgcn_s_wait_asynccnt(0); }
#else
DEV void async_wait0() {}
#endif

// Build a WMMA 16-bit A/B operand fragment from LDS.
// Layout (ISA 7.12.2, 16-bit A 16x32): lane (g = lane>>4, r = lane&15) holds
// row r, K-pairs {g*8 .. g*8+7} in VGPRs 0-3 and {16+g*8 ..} in VGPRs 4-7,
// i.e. two contiguous 16-byte chunks -> two ds_load_b128.
DEV v16h ld_frag(const f16* p) {
  v8h lo = *(const v8h*)p;
  v8h hi = *(const v8h*)(p + 16);
  v16h r;
#pragma unroll
  for (int i = 0; i < 8; ++i) { r[i] = lo[i]; r[i + 8] = hi[i]; }
  return r;
}

DEV float gelu_f(float x) {
  float x3 = x * x * x;
  return 0.5f * x * (1.0f + tanhf(0.7978845608f * (x + 0.044715f * x3)));
}

// ---------------------------------------------------------------------------
// Generic f16 WMMA GEMM: C[M,N] = A[M,K] @ Bt^T + bias (+resid) (+gelu)
// A row-major [M][K]; Bt is the weight pre-transposed to N-major [N][K].
// Block tile 64x128, 8 waves, ping-pong LDS double buffering, async DMA
// global->LDS when available. M % 64 == 0, K % 32 == 0; N masked at store.
// ---------------------------------------------------------------------------
template<bool OUTF16, bool GELU, bool RESID>
__global__ __launch_bounds__(256)
void gemm_f16(const f16* __restrict__ A, const f16* __restrict__ Bt,
              const float* __restrict__ bias, const float* __restrict__ resid,
              void* __restrict__ Cout, int M, int N, int K)
{
  __shared__ __align__(16) f16 Asm[2][64 * 32];    // [buf][m][k]
  __shared__ __align__(16) f16 Bsm[2][128 * 32];   // [buf][n][k]

  const int tid  = threadIdx.x;
  const int wv   = tid >> 5, lane = tid & 31;
  const int mt   = wv & 3;          // M-tile of this wave (0..3)
  const int nq   = wv >> 2;         // N-quad (0..1) -> n-tiles nq*4..nq*4+3
  const int g    = lane >> 4, lr = lane & 15;
  const int m0   = blockIdx.y * 64, n0 = blockIdx.x * 128;

  const int ar = tid >> 2;          // A loader: row 0..63
  const int ac = (tid & 3) * 8;     // A loader: k chunk (8 f16)
  const int br = tid >> 1;          // B loader: row 0..127
  const int bc = (tid & 1) * 16;    // B loader: k chunk (16 f16)
  int nr = n0 + br; if (nr > N - 1) nr = N - 1;   // clamp; masked at store

  const f16* agp = A  + (size_t)(m0 + ar) * K + ac;
  const f16* bgp = Bt + (size_t)nr * K + bc;

  v8f acc[4] = {};

  // prologue: fill buffer 0
  {
#if USE_ASYNC_LDS
    async_cp16(agp,     &Asm[0][ar * 32 + ac]);
    async_cp16(bgp,     &Bsm[0][br * 32 + bc]);
    async_cp16(bgp + 8, &Bsm[0][br * 32 + bc + 8]);
#else
    *(uint4*)&Asm[0][ar * 32 + ac]     = *(const uint4*)agp;
    *(uint4*)&Bsm[0][br * 32 + bc]     = *(const uint4*)bgp;
    *(uint4*)&Bsm[0][br * 32 + bc + 8] = *(const uint4*)(bgp + 8);
#endif
  }
  async_wait0();
  __syncthreads();

  int cur = 0;
  for (int k0 = 0; k0 < K; k0 += 32) {
    if (k0 + 32 < K) {           // issue next tile's DMA before computing
      int nx = cur ^ 1;
#if USE_ASYNC_LDS
      async_cp16(agp + k0 + 32,     &Asm[nx][ar * 32 + ac]);
      async_cp16(bgp + k0 + 32,     &Bsm[nx][br * 32 + bc]);
      async_cp16(bgp + k0 + 40,     &Bsm[nx][br * 32 + bc + 8]);
#else
      *(uint4*)&Asm[nx][ar * 32 + ac]     = *(const uint4*)(agp + k0 + 32);
      *(uint4*)&Bsm[nx][br * 32 + bc]     = *(const uint4*)(bgp + k0 + 32);
      *(uint4*)&Bsm[nx][br * 32 + bc + 8] = *(const uint4*)(bgp + k0 + 40);
#endif
    }
    v16h a = ld_frag(&Asm[cur][(mt * 16 + lr) * 32 + g * 8]);
#pragma unroll
    for (int t = 0; t < 4; ++t) {
      v16h b = ld_frag(&Bsm[cur][((nq * 4 + t) * 16 + lr) * 32 + g * 8]);
      acc[t] = __builtin_amdgcn_wmma_f32_16x16x32_f16(
          false, a, false, b, (short)0, acc[t], false, false);
    }
    async_wait0();
    __syncthreads();
    cur ^= 1;
  }

  // Epilogue. D layout: lane holds column n = tile*16 + lr, rows
  // m = mt*16 + rr + 8*g for rr=0..7.
#pragma unroll
  for (int t = 0; t < 4; ++t) {
    int n = n0 + (nq * 4 + t) * 16 + lr;
    if (n < N) {
      float bv = bias ? bias[n] : 0.f;
#pragma unroll
      for (int rr = 0; rr < 8; ++rr) {
        int m = m0 + mt * 16 + rr + 8 * g;
        float v = acc[t][rr] + bv;
        if (RESID) v += resid[(size_t)m * N + n];
        if (GELU)  v = gelu_f(v);
        if (OUTF16) ((f16*)Cout)[(size_t)m * N + n] = (f16)v;
        else        ((float*)Cout)[(size_t)m * N + n] = v;
      }
    }
  }
}

// ---------------------------------------------------------------------------
// Fused windowed attention, one block per (stream, batch, window, head).
// LDS: K[256x96 pad] 48K, Q[64x96 pad] 12K, V^T[80x256 pad] 40K, scratch 1K.
// S = QK^T (WMMA), shfl+LDS softmax, P overwrites K region, O = P V (WMMA).
// ---------------------------------------------------------------------------
__global__ __launch_bounds__(256)
void attn_kernel(const f16* __restrict__ qkv, f16* __restrict__ o_out)
{
  __shared__ __align__(16) f16 sK[256 * 96];
  __shared__ __align__(16) f16 sQ[64 * 96];
  __shared__ __align__(16) f16 sVt[80 * 256];
  __shared__ float sred[2][4][2][8][2];   // [max/sum][mt][half][row][nh]

  const int bid  = blockIdx.x;
  const int head = bid & 7;
  const int win  = (bid >> 3) & 15;
  const int bb   = (bid >> 7) & 7;
  const int si   = bid >> 10;
  const int wy = win >> 2, wx = win & 3;
  const int tid = threadIdx.x;
  const size_t imgbase = (size_t)(si * 8 + bb) * 64 * 64;
  const int qc = head * 72, kc = 576 + head * 72, vc = 1152 + head * 72;

  // zero the pads
  for (int i = tid; i < 256 * 24; i += 256) { int r = i / 24; sK[r * 96 + 72 + i % 24] = (f16)0.f; }
  for (int i = tid; i < 64 * 24;  i += 256) { int r = i / 24; sQ[r * 96 + 72 + i % 24] = (f16)0.f; }
  for (int i = tid; i < 8 * 256;  i += 256) { sVt[72 * 256 + i] = (f16)0.f; }

  // K: 256 keys x 72, row-major padded to 96 (b64 chunks)
  for (int i = tid; i < 256 * 18; i += 256) {
    int row = i / 18, c4 = (i % 18) * 4;
    size_t tok = imgbase + (size_t)(wy * 16 + (row >> 4)) * 64 + wx * 16 + (row & 15);
    *(uint2*)&sK[row * 96 + c4] = *(const uint2*)(qkv + tok * 1728 + kc + c4);
  }
  // V transposed: Vt[d][key]
  for (int i = tid; i < 72 * 256; i += 256) {
    int d = i >> 8, key = i & 255;
    size_t tok = imgbase + (size_t)(wy * 16 + (key >> 4)) * 64 + wx * 16 + (key & 15);
    sVt[d * 256 + key] = qkv[tok * 1728 + vc + d];
  }
  // Q: 2x2 max-pooled, pre-scaled by 1/sqrt(72)
  const float qscale = 0.117851130f;
  for (int i = tid; i < 64 * 72; i += 256) {
    int p = i / 72, d = i % 72;
    int py = p >> 3, px = p & 7;
    float mx = -1e30f;
#pragma unroll
    for (int dy = 0; dy < 2; ++dy)
#pragma unroll
      for (int dx = 0; dx < 2; ++dx) {
        size_t tok = imgbase + (size_t)(wy * 16 + py * 2 + dy) * 64 + wx * 16 + px * 2 + dx;
        mx = fmaxf(mx, (float)qkv[tok * 1728 + qc + d]);
      }
    sQ[p * 96 + d] = (f16)(mx * qscale);
  }
  __syncthreads();

  const int wv = tid >> 5, lane = tid & 31;
  const int mt = wv & 3, nh = wv >> 2;      // 4 M-tiles x 2 key-halves
  const int g = lane >> 4, lr = lane & 15;

  // S = Q K^T : M=64, N=256, K=96
  v8f s[8] = {};
#pragma unroll
  for (int ks = 0; ks < 96; ks += 32) {
    v16h a = ld_frag(&sQ[(mt * 16 + lr) * 96 + ks + g * 8]);
#pragma unroll
    for (int t = 0; t < 8; ++t) {
      v16h b = ld_frag(&sK[((nh * 8 + t) * 16 + lr) * 96 + ks + g * 8]);
      s[t] = __builtin_amdgcn_wmma_f32_16x16x32_f16(
          false, a, false, b, (short)0, s[t], false, false);
    }
  }

  // row-wise softmax over 256 keys
  float rmax[8];
#pragma unroll
  for (int rr = 0; rr < 8; ++rr) {
    float v = -1e30f;
#pragma unroll
    for (int t = 0; t < 8; ++t) v = fmaxf(v, s[t][rr]);
#pragma unroll
    for (int off = 1; off < 16; off <<= 1) v = fmaxf(v, __shfl_xor(v, off, 32));
    rmax[rr] = v;
  }
  if (lr == 0) {
#pragma unroll
    for (int rr = 0; rr < 8; ++rr) sred[0][mt][g][rr][nh] = rmax[rr];
  }
  __syncthreads();
#pragma unroll
  for (int rr = 0; rr < 8; ++rr)
    rmax[rr] = fmaxf(sred[0][mt][g][rr][0], sred[0][mt][g][rr][1]);
  float rsum[8];
#pragma unroll
  for (int rr = 0; rr < 8; ++rr) {
    float sum = 0.f;
#pragma unroll
    for (int t = 0; t < 8; ++t) { float e = __expf(s[t][rr] - rmax[rr]); s[t][rr] = e; sum += e; }
#pragma unroll
    for (int off = 1; off < 16; off <<= 1) sum += __shfl_xor(sum, off, 32);
    rsum[rr] = sum;
  }
  if (lr == 0) {
#pragma unroll
    for (int rr = 0; rr < 8; ++rr) sred[1][mt][g][rr][nh] = rsum[rr];
  }
  __syncthreads();   // all waves done reading sK -> safe to overwrite with P

  f16* sP = sK;      // P[64][256] row-major, reuses K region
#pragma unroll
  for (int rr = 0; rr < 8; ++rr) {
    float inv = 1.f / (sred[1][mt][g][rr][0] + sred[1][mt][g][rr][1]);
    int m = mt * 16 + rr + 8 * g;
#pragma unroll
    for (int t = 0; t < 8; ++t)
      sP[m * 256 + (nh * 8 + t) * 16 + lr] = (f16)(s[t][rr] * inv);
  }
  __syncthreads();

  // O = P V : M=64, N=72 (5 n-tiles, last masked), K=256
  const int ntbase = (nh == 0) ? 0 : 3;
  const int ntcnt  = (nh == 0) ? 3 : 2;
  v8f o[3] = {};
  for (int ks = 0; ks < 256; ks += 32) {
    v16h a = ld_frag(&sP[(mt * 16 + lr) * 256 + ks + g * 8]);
    for (int j = 0; j < ntcnt; ++j) {
      v16h b = ld_frag(&sVt[((ntbase + j) * 16 + lr) * 256 + ks + g * 8]);
      o[j] = __builtin_amdgcn_wmma_f32_16x16x32_f16(
          false, a, false, b, (short)0, o[j], false, false);
    }
  }
  // store straight into the un-windowed (B,32,32,576) raster
  for (int j = 0; j < ntcnt; ++j) {
    int nn = (ntbase + j) * 16 + lr;
    if (nn < 72) {
      int col = head * 72 + nn;
#pragma unroll
      for (int rr = 0; rr < 8; ++rr) {
        int m = mt * 16 + rr + 8 * g;
        int hh = wy * 8 + (m >> 3), ww = wx * 8 + (m & 7);
        size_t tok = ((size_t)(si * 8 + bb) * 32 + hh) * 32 + ww;
        o_out[tok * 576 + col] = (f16)o[j][rr];
      }
    }
  }
}

// ---------------------------------------------------------------------------
// LayerNorm (one wave32 per token), optional raw f16 cast output
// ---------------------------------------------------------------------------
__global__ __launch_bounds__(256)
void ln_kernel(const float* __restrict__ x, const float* __restrict__ gma,
               const float* __restrict__ bta, f16* __restrict__ out,
               f16* __restrict__ xf16, int D, int ntok)
{
  int tok = blockIdx.x * 8 + (threadIdx.x >> 5);
  int lane = threadIdx.x & 31;
  if (tok >= ntok) return;
  const float* xr = x + (size_t)tok * D;
  float s = 0.f, s2 = 0.f;
  for (int c = lane; c < D; c += 32) { float v = xr[c]; s += v; s2 += v * v; }
#pragma unroll
  for (int off = 1; off < 32; off <<= 1) {
    s += __shfl_xor(s, off, 32); s2 += __shfl_xor(s2, off, 32);
  }
  float mean = s / D;
  float rstd = rsqrtf(s2 / D - mean * mean + 1e-6f);
  f16* orow = out + (size_t)tok * D;
  for (int c = lane; c < D; c += 32) {
    float v = xr[c];
    orow[c] = (f16)((v - mean) * rstd * gma[c] + bta[c]);
    if (xf16) xf16[(size_t)tok * D + c] = (f16)v;
  }
}

// transpose-convert f32 [K][N] -> f16 N-major [N][K]
__global__ void cvt_t_kernel(const float* __restrict__ src, f16* __restrict__ dst,
                             int K, int N)
{
  for (int i = blockIdx.x * blockDim.x + threadIdx.x; i < N * K;
       i += gridDim.x * blockDim.x) {
    int n = i / K, k = i % K;
    dst[i] = (f16)src[(size_t)k * N + n];
  }
}

// 2x2 max-pool of full-res proj output -> f32 shortcut (pooled raster)
__global__ void pool_kernel(const f16* __restrict__ pf, float* __restrict__ sc)
{
  for (size_t i = (size_t)blockIdx.x * blockDim.x + threadIdx.x;
       i < (size_t)16384 * 576; i += (size_t)gridDim.x * blockDim.x) {
    int c = (int)(i % 576); size_t t = i / 576;
    int ww = (int)(t % 32); int hh = (int)((t / 32) % 32); int img = (int)(t / 1024);
    size_t base = ((size_t)img * 64 + hh * 2) * 64 + ww * 2;
    float v0 = pf[(base)      * 576 + c], v1 = pf[(base + 1)  * 576 + c];
    float v2 = pf[(base + 64) * 576 + c], v3 = pf[(base + 65) * 576 + c];
    sc[i] = fmaxf(fmaxf(v0, v1), fmaxf(v2, v3));
  }
}

// out = x + mlp + 0.5 * (g0*y0 + g1*y1), gate = softmax(x @ gw + gb)
__global__ __launch_bounds__(256)
void final_kernel(const float* __restrict__ x, const float* __restrict__ mlp2,
                  const float* __restrict__ y, const float* __restrict__ gw,
                  const float* __restrict__ gb, float* __restrict__ out)
{
  int tok = blockIdx.x * 8 + (threadIdx.x >> 5);
  int lane = threadIdx.x & 31;
  int si = tok >> 13, tl = tok & 8191;
  const float* xr = x + (size_t)tok * 576;
  float l0 = 0.f, l1 = 0.f;
  for (int c = lane; c < 576; c += 32) {
    float v = xr[c];
    l0 += v * gw[si * 1152 + c * 2 + 0];
    l1 += v * gw[si * 1152 + c * 2 + 1];
  }
#pragma unroll
  for (int off = 1; off < 32; off <<= 1) {
    l0 += __shfl_xor(l0, off, 32); l1 += __shfl_xor(l1, off, 32);
  }
  l0 += gb[si * 2 + 0]; l1 += gb[si * 2 + 1];
  float m = fmaxf(l0, l1);
  float e0 = __expf(l0 - m), e1 = __expf(l1 - m);
  float inv = 0.5f / (e0 + e1);
  float g0 = e0 * inv, g1 = e1 * inv;
  const float* y0 = y + ((size_t)(si * 2 + 0) * 8192 + tl) * 576;
  const float* y1 = y + ((size_t)(si * 2 + 1) * 8192 + tl) * 576;
  const float* mr = mlp2 + (size_t)tok * 576;
  float* orow = out + (size_t)tok * 576;
  for (int c = lane; c < 576; c += 32)
    orow[c] = xr[c] + mr[c] + g0 * y0[c] + g1 * y1[c];
}

// ---------------------------------------------------------------------------
extern "C" void kernel_launch(void* const* d_in, const int* in_sizes, int n_in,
                              void* d_out, int out_size, void* d_ws, size_t ws_size,
                              hipStream_t stream)
{
  (void)in_sizes; (void)n_in; (void)out_size; (void)ws_size;
  const float* rgb    = (const float*)d_in[0];
  const float* depth  = (const float*)d_in[1];
  const float* n1g    = (const float*)d_in[2];
  const float* n1b    = (const float*)d_in[3];
  const float* proj_w = (const float*)d_in[4];
  const float* proj_b = (const float*)d_in[5];
  const float* qkv_w  = (const float*)d_in[6];
  const float* qkv_b  = (const float*)d_in[7];
  const float* apw    = (const float*)d_in[8];
  const float* apb    = (const float*)d_in[9];
  const float* n2g    = (const float*)d_in[10];
  const float* n2b    = (const float*)d_in[11];
  const float* mw1    = (const float*)d_in[12];
  const float* mb1    = (const float*)d_in[13];
  const float* mw2    = (const float*)d_in[14];
  const float* mb2    = (const float*)d_in[15];
  const float* gw     = (const float*)d_in[16];
  const float* gb     = (const float*)d_in[17];
  const float* dw     = (const float*)d_in[18];
  const float* db     = (const float*)d_in[19];
  const float* uw     = (const float*)d_in[20];
  const float* ub     = (const float*)d_in[21];

  char* ws = (char*)d_ws;
  size_t off = 0;
  auto alloc = [&](size_t bytes) -> char* {
    char* p = ws + off;
    off += (bytes + 255) & ~(size_t)255;
    return p;
  };

  // f16 weights, pre-transposed to N-major [N][K]
  f16* w_proj = (f16*)alloc((size_t)576 * 288 * 2);
  f16* w_qkv  = (f16*)alloc((size_t)1728 * 288 * 2);
  f16* w_ap   = (f16*)alloc((size_t)576 * 576 * 2);
  f16* w_m1   = (f16*)alloc((size_t)2304 * 576 * 2);
  f16* w_m2   = (f16*)alloc((size_t)576 * 2304 * 2);
  f16* w_dn   = (f16*)alloc((size_t)4 * 32 * 576 * 2);
  f16* w_up   = (f16*)alloc((size_t)4 * 576 * 32 * 2);
  // activations (with lifetime-based aliasing)
  f16*   xn       = (f16*)alloc((size_t)65536 * 288 * 2);      // LN1 out
  char*  qkv_reg  = alloc((size_t)65536 * 1728 * 2);           // qkv f16
  char*  pf_reg   = alloc((size_t)65536 * 576 * 2);            // proj full-res f16
  float* shortcut = (float*)alloc((size_t)16384 * 576 * 4);
  float* xbuf     = (float*)alloc((size_t)16384 * 576 * 4);    // x after attn
  f16*   adh      = (f16*)alloc((size_t)4 * 8192 * 32 * 2);    // adapter hidden
  f16* qkvb   = (f16*)qkv_reg;
  f16* mlp1   = (f16*)qkv_reg;                                  // reuse after attention
  float* ybuf = (float*)(qkv_reg + (size_t)16384 * 2304 * 2);   // adapter up outs
  f16* pfull  = (f16*)pf_reg;
  f16* o16    = (f16*)pf_reg;                                   // reuse after pool
  f16* x16    = (f16*)(pf_reg + (size_t)16384 * 576 * 2);
  f16* h2     = (f16*)(pf_reg + (size_t)2 * 16384 * 576 * 2);
  float* mlp2 = (float*)xn;                                     // reuse after qkv GEMM

  // 1) weight conversion (f32 [K][N] -> f16 [N][K])
  cvt_t_kernel<<<256, 256, 0, stream>>>(proj_w, w_proj, 288, 576);
  cvt_t_kernel<<<256, 256, 0, stream>>>(qkv_w,  w_qkv,  288, 1728);
  cvt_t_kernel<<<256, 256, 0, stream>>>(apw,    w_ap,   576, 576);
  cvt_t_kernel<<<256, 256, 0, stream>>>(mw1,    w_m1,   576, 2304);
  cvt_t_kernel<<<256, 256, 0, stream>>>(mw2,    w_m2,   2304, 576);
  for (int idx = 0; idx < 4; ++idx) {
    cvt_t_kernel<<<32, 256, 0, stream>>>(dw + (size_t)idx * 576 * 32,
                                         w_dn + (size_t)idx * 32 * 576, 576, 32);
    cvt_t_kernel<<<32, 256, 0, stream>>>(uw + (size_t)idx * 32 * 576,
                                         w_up + (size_t)idx * 576 * 32, 32, 576);
  }

  // 2) LN1 for both streams -> xn f16
  ln_kernel<<<4096, 256, 0, stream>>>(rgb,   n1g, n1b, xn,                       nullptr, 288, 32768);
  ln_kernel<<<4096, 256, 0, stream>>>(depth, n1g, n1b, xn + (size_t)32768 * 288, nullptr, 288, 32768);

  // 3) proj GEMM (full res) then 2x2 pool -> shortcut
  gemm_f16<true, false, false><<<dim3(5, 1024), 256, 0, stream>>>(
      xn, w_proj, proj_b, nullptr, pfull, 65536, 576, 288);
  pool_kernel<<<4096, 256, 0, stream>>>(pfull, shortcut);

  // 4) qkv GEMM
  gemm_f16<true, false, false><<<dim3(14, 1024), 256, 0, stream>>>(
      xn, w_qkv, qkv_b, nullptr, qkvb, 65536, 1728, 288);

  // 5) fused windowed attention -> o16 (unwindowed raster)
  attn_kernel<<<2048, 256, 0, stream>>>(qkvb, o16);

  // 6) attn projection + pooled shortcut residual -> x (f32)
  gemm_f16<false, false, true><<<dim3(5, 256), 256, 0, stream>>>(
      o16, w_ap, apb, shortcut, xbuf, 16384, 576, 576);

  // 7) LN2 -> h2 f16 (+ raw x f16 for the adapter)
  ln_kernel<<<2048, 256, 0, stream>>>(xbuf, n2g, n2b, h2, x16, 576, 16384);

  // 8) MLP
  gemm_f16<true, true, false><<<dim3(18, 256), 256, 0, stream>>>(
      h2, w_m1, mb1, nullptr, mlp1, 16384, 2304, 576);
  gemm_f16<false, false, false><<<dim3(5, 256), 256, 0, stream>>>(
      mlp1, w_m2, mb2, nullptr, mlp2, 16384, 576, 2304);

  // 9) adapter: per (stream, expert) down(gelu) then up
  for (int si = 0; si < 2; ++si)
    for (int e = 0; e < 2; ++e) {
      int idx = si * 2 + e;
      gemm_f16<true, true, false><<<dim3(1, 128), 256, 0, stream>>>(
          x16 + (size_t)si * 8192 * 576, w_dn + (size_t)idx * 32 * 576,
          db + idx * 32, nullptr, adh + (size_t)idx * 8192 * 32,
          8192, 32, 576);
      gemm_f16<false, false, false><<<dim3(5, 128), 256, 0, stream>>>(
          adh + (size_t)idx * 8192 * 32, w_up + (size_t)idx * 576 * 32,
          ub + idx * 576, nullptr, ybuf + (size_t)idx * 8192 * 576,
          8192, 576, 32);
    }

  // 10) gate + combine -> d_out (f32)
  final_kernel<<<2048, 256, 0, stream>>>(xbuf, mlp2, ybuf, gw, gb, (float*)d_out);
}
